// SelfAttention_61718680043981
// MI455X (gfx1250) — compile-verified, matrix operand order checked by reading, not run on hardware
//
#include <hip/hip_runtime.h>
#include <hip/hip_bf16.h>

typedef __bf16 bf16;
typedef __attribute__((ext_vector_type(16))) __bf16 v16bf;
typedef __attribute__((ext_vector_type(8)))  __bf16 bf16x8;
typedef __attribute__((ext_vector_type(8)))  float  v8f;

#define LDSK 40  // padded LDS row stride (bf16 elems) for the 32-wide K tile

// ---- optional gfx1250 async-to-LDS path (guarded; falls back to sync) ------
#if defined(__has_builtin)
#  if __has_builtin(__builtin_amdgcn_global_load_async_to_lds_b128)
#    define HAVE_ASYNC_LDS 1
#  endif
#endif
#ifndef HAVE_ASYNC_LDS
#  define HAVE_ASYNC_LDS 0
#endif

#if HAVE_ASYNC_LDS
// Exact param types per hipcc diagnostic: non-const pointer to
// 'int __attribute__((vector_size(16)))' in AS(1) (global) / AS(3) (LDS).
typedef int v4i_ __attribute__((vector_size(16)));
// 16B global->LDS async copy, one transfer per lane. Flat LDS pointers keep
// the LDS offset in the low 32 bits (ISA 10.2), so truncation is correct.
__device__ __forceinline__ void async_cp16(const bf16* g, bf16* l) {
    __builtin_amdgcn_global_load_async_to_lds_b128(
        (__attribute__((address_space(1))) v4i_*)(uintptr_t)g,
        (__attribute__((address_space(3))) v4i_*)(unsigned)(uintptr_t)l,
        0, 0);
}
__device__ __forceinline__ void async_wait0() {
#  if __has_builtin(__builtin_amdgcn_s_wait_asynccnt)
    __builtin_amdgcn_s_wait_asynccnt(0);
#  else
    asm volatile("s_wait_asynccnt 0x0" ::: "memory");
#  endif
}
#endif

// Load one 16x32 bf16 WMMA operand fragment from an LDS tile (row-major,
// stride LDSK). ISA 7.12.2 layout: lanes 0-15 hold K 0..7 & 16..23,
// lanes 16-31 hold K 8..15 & 24..31; row = base + lane%16.
__device__ __forceinline__ v16bf load_frag(const bf16* t, int row, int lane) {
    const int kb = (lane < 16) ? 0 : 8;
    union { v16bf v; bf16x8 h[2]; } u;
    u.h[0] = *(const bf16x8*)(t + row * LDSK + kb);
    u.h[1] = *(const bf16x8*)(t + row * LDSK + kb + 16);
    return u.v;
}

// C[m,n] = sum_k A[m,k] * B[n,k]  ("NT" GEMM), bf16 in, fp32 accumulate.
// Block tile 128x256, K-tile 32. 256 threads = 8 waves (2 in M x 4 in N),
// each wave owns a 64x64 tile: 4 A-frags + 4 B-frags -> 16 WMMAs per K-step.
// Requires M % 128 == 0, N % 256 == 0, K % 32 == 0 (true for all launches).
template<int OUT_BF16>
__global__ __launch_bounds__(256)
void gemm_nt_bf16(const bf16* __restrict__ A, const bf16* __restrict__ B,
                  void* __restrict__ Cv,
                  int M, int N, int K, int lda, int ldb, int ldc,
                  long long strideA, long long strideB, long long strideC)
{
    __shared__ bf16 ldsA[128 * LDSK];
    __shared__ bf16 ldsB[256 * LDSK];

    const int zb = blockIdx.z;
    A += (long long)zb * strideA;
    B += (long long)zb * strideB;

    const int m0   = blockIdx.y * 128;
    const int n0   = blockIdx.x * 256;
    const int tid  = threadIdx.x;
    const int lane = tid & 31;
    const int wave = tid >> 5;
    const int wm   = (wave & 1) * 64;   // 2 waves along M
    const int wn   = (wave >> 1) * 64;  // 4 waves along N

    const int ar = tid >> 1;            // A stage: row 0..127
    const int as = (tid & 1) * 16;      // A stage: 16-elem K segment

    v8f acc[4][4] = {};

    for (int kk = 0; kk < K; kk += 32) {
        const bf16* gA = A + (long long)(m0 + ar) * lda + kk + as;
        const bf16* gB = B + (long long)(n0 + tid) * ldb + kk;  // full 32-elem row
        bf16* lA = &ldsA[ar * LDSK + as];
        bf16* lB = &ldsB[tid * LDSK];
#if HAVE_ASYNC_LDS
        async_cp16(gA,      lA);
        async_cp16(gA + 8,  lA + 8);
#pragma unroll
        for (int q = 0; q < 4; ++q)
            async_cp16(gB + 8 * q, lB + 8 * q);
#else
        {
            const uint4* pa = (const uint4*)gA;
            uint4 a0 = pa[0], a1 = pa[1];
            const uint4* pb = (const uint4*)gB;
            uint4 b0 = pb[0], b1 = pb[1], b2 = pb[2], b3 = pb[3];
            uint4* qa = (uint4*)lA;
            qa[0] = a0; qa[1] = a1;
            uint4* qb = (uint4*)lB;
            qb[0] = b0; qb[1] = b1; qb[2] = b2; qb[3] = b3;
        }
#endif
        if (kk + 32 < K) {  // pull next K-tile toward the caches
            __builtin_prefetch(gA + 32, 0, 1);
            __builtin_prefetch(gB + 32, 0, 1);
        }
#if HAVE_ASYNC_LDS
        async_wait0();
#endif
        __syncthreads();

        v16bf af[4];
#pragma unroll
        for (int i = 0; i < 4; ++i)
            af[i] = load_frag(ldsA, wm + i * 16 + (lane & 15), lane);
        v16bf bfv[4];
#pragma unroll
        for (int j = 0; j < 4; ++j)
            bfv[j] = load_frag(ldsB, wn + j * 16 + (lane & 15), lane);

#pragma unroll
        for (int i = 0; i < 4; ++i)
#pragma unroll
            for (int j = 0; j < 4; ++j)
                acc[i][j] = __builtin_amdgcn_wmma_f32_16x16x32_bf16(
                    false, af[i], false, bfv[j], (short)0, acc[i][j], false, false);

        __syncthreads();
    }

    // Store per ISA C/D layout: VGPR r -> M = r + (lane>=16 ? 8 : 0), N = lane%16.
    const int nl = lane & 15;
    const int mh = (lane >> 4) * 8;
    const long long cbase = (long long)zb * strideC;
#pragma unroll
    for (int i = 0; i < 4; ++i) {
#pragma unroll
        for (int j = 0; j < 4; ++j) {
            const int gm = m0 + wm + i * 16 + mh;
            const int gn = n0 + wn + j * 16 + nl;
#pragma unroll
            for (int r = 0; r < 8; ++r) {
                const float v = acc[i][j][r];
                const long long idx = cbase + (long long)(gm + r) * ldc + gn;
                if (OUT_BF16) ((bf16*)Cv)[idx] = (bf16)v;
                else          ((float*)Cv)[idx] = v;
            }
        }
    }
}

__global__ void cvt_scale_bf16(const float* __restrict__ s, bf16* __restrict__ d,
                               long long n, float scale)
{
    long long i  = (long long)blockIdx.x * blockDim.x + threadIdx.x;
    long long st = (long long)gridDim.x * blockDim.x;
    for (; i < n; i += st) d[i] = (bf16)(s[i] * scale);
}

// dst[c, r] = src[r, c]   (bf16, per batch), 32x32 LDS tiles.
__global__ __launch_bounds__(256)
void transpose_bf16(const bf16* __restrict__ src, bf16* __restrict__ dst,
                    int rows, int cols, long long sS, long long sD)
{
    __shared__ bf16 tile[32][33];
    const bf16* s  = src + (long long)blockIdx.z * sS;
    bf16*       dd = dst + (long long)blockIdx.z * sD;
    const int tx = threadIdx.x & 31;
    const int ty = threadIdx.x >> 5;      // 0..7
    const int x0 = blockIdx.x * 32;       // col block in src
    const int y0 = blockIdx.y * 32;       // row block in src
#pragma unroll
    for (int i = 0; i < 4; ++i)
        tile[ty + i * 8][tx] = s[(long long)(y0 + ty + i * 8) * cols + x0 + tx];
    __syncthreads();
#pragma unroll
    for (int i = 0; i < 4; ++i)
        dd[(long long)(x0 + ty + i * 8) * rows + y0 + tx] = tile[tx][ty + i * 8];
}

// Row-wise masked softmax of M1 (n==1024, 256 thr x 4 elems).
// M1[b,i,j] = logits[j,i], so row-softmax of M1 == reference softmax(axis=1).
// Writes P (bf16, row-major, feeds P@V) and attention weights transposed into
// d_out: aw[b,j,i] = P[b,i,j].
__global__ __launch_bounds__(256)
void softmax_rows(const float* __restrict__ L, bf16* __restrict__ P,
                  float* __restrict__ AW, int n)
{
    const int i   = blockIdx.x;
    const int b   = blockIdx.y;
    const int tid = threadIdx.x;
    const float* row = L + ((long long)b * n + i) * n;

    float v[4];
    float mx = -3.4e38f;
#pragma unroll
    for (int t = 0; t < 4; ++t) {
        const int j = tid + t * 256;
        float x = row[j];
        if (j == i) x = -__builtin_inff();   // diagonal self-exclusion
        v[t] = x;
        mx = fmaxf(mx, x);
    }
    __shared__ float red[256];
    red[tid] = mx; __syncthreads();
    for (int s = 128; s > 0; s >>= 1) {
        if (tid < s) red[tid] = fmaxf(red[tid], red[tid + s]);
        __syncthreads();
    }
    mx = red[0]; __syncthreads();

    float sum = 0.f;
#pragma unroll
    for (int t = 0; t < 4; ++t) {
        const float e = __expf(v[t] - mx);   // exp(-inf)=0 kills the diagonal
        v[t] = e; sum += e;
    }
    red[tid] = sum; __syncthreads();
    for (int s = 128; s > 0; s >>= 1) {
        if (tid < s) red[tid] += red[tid + s];
        __syncthreads();
    }
    const float inv = 1.f / red[0];

    bf16*  prow = P  + ((long long)b * n + i) * n;
    float* awb  = AW + (long long)b * n * n;
#pragma unroll
    for (int t = 0; t < 4; ++t) {
        const int j = tid + t * 256;
        const float p = v[t] * inv;
        prow[j] = (bf16)p;
        awb[(long long)j * n + i] = p;       // transposed write -> aw[b,j,i]
    }
}

extern "C" void kernel_launch(void* const* d_in, const int* in_sizes, int n_in,
                              void* d_out, int out_size, void* d_ws, size_t ws_size,
                              hipStream_t stream)
{
    (void)in_sizes; (void)n_in; (void)out_size; (void)ws_size;
    const float* y  = (const float*)d_in[0];
    const float* Wk = (const float*)d_in[1];
    const float* Wq = (const float*)d_in[2];
    const float* Wv = (const float*)d_in[3];
    const float* Wo = (const float*)d_in[4];

    const int B = 8, N = 1024, D = 2048;
    const long long YN = (long long)B * N * D;  // 16,777,216
    const long long WN = (long long)D * D;      //  4,194,304
    const long long LN = (long long)B * N * N;  //  8,388,608

    char* ws = (char*)d_ws;
    size_t off = 0;
    auto take = [&](size_t bytes) {
        void* p = ws + off;
        off += (bytes + 255) & ~(size_t)255;
        return p;
    };

    bf16* yb  = (bf16*)take(YN * 2);  // bf16 y; later reused as fp32 M1 (same bytes)
    bf16* Wkb = (bf16*)take(WN * 2);
    bf16* Wqb = (bf16*)take(WN * 2);  // carries the 0.1 Q-scale
    bf16* Wvb = (bf16*)take(WN * 2);
    bf16* Wob = (bf16*)take(WN * 2);
    bf16* Kb  = (bf16*)take(YN * 2);  // later reused as out_pre (bf16)
    bf16* Qb  = (bf16*)take(YN * 2);  // later reused as V^T (bf16)
    bf16* Vb  = (bf16*)take(YN * 2);
    bf16* Pb  = (bf16*)take(LN * 2);

    float* M1 = (float*)yb;           // [B, N, N] fp32 logits^T, aliases yb
    bf16*  Vt = Qb;                   // [B, D, N] bf16, aliases Qb
    bf16*  Ob = Kb;                   // [B*N, D]  bf16 out_pre, aliases Kb

    const dim3 blk(256);

    // fp32 -> bf16 conversions (0.1 folded into Wq)
    cvt_scale_bf16<<<dim3(1024), blk, 0, stream>>>(y,  yb,  YN, 1.0f);
    cvt_scale_bf16<<<dim3(512),  blk, 0, stream>>>(Wk, Wkb, WN, 1.0f);
    cvt_scale_bf16<<<dim3(512),  blk, 0, stream>>>(Wq, Wqb, WN, 0.1f);
    cvt_scale_bf16<<<dim3(512),  blk, 0, stream>>>(Wv, Wvb, WN, 1.0f);
    cvt_scale_bf16<<<dim3(512),  blk, 0, stream>>>(Wo, Wob, WN, 1.0f);

    // K/Q/V projections: [8192 x 2048] = y(bf16) @ W^T   (bf16 out)
    const dim3 gproj(D / 256, (B * N) / 128, 1);
    gemm_nt_bf16<1><<<gproj, blk, 0, stream>>>(yb, Wkb, Kb, B * N, D, D, D, D, D, 0, 0, 0);
    gemm_nt_bf16<1><<<gproj, blk, 0, stream>>>(yb, Wqb, Qb, B * N, D, D, D, D, D, 0, 0, 0);
    gemm_nt_bf16<1><<<gproj, blk, 0, stream>>>(yb, Wvb, Vb, B * N, D, D, D, D, D, 0, 0, 0);

    // M1[b,i,j] = K_i . Q_j  (= logits[j,i]); fp32 out, overwrites yb
    const dim3 glog(N / 256, N / 128, B);
    gemm_nt_bf16<0><<<glog, blk, 0, stream>>>(Kb, Qb, M1, N, N, D, D, D, N,
                                              (long long)N * D, (long long)N * D,
                                              (long long)N * N);

    // Row softmax of M1 -> Pb (bf16) and attention weights (fp32, d_out tail)
    float* out_f = (float*)d_out;
    float* aw    = out_f + YN;
    softmax_rows<<<dim3(N, B), blk, 0, stream>>>(M1, Pb, aw, N);

    // Vt[b,d,j] = V[b,j,d]
    transpose_bf16<<<dim3(D / 32, N / 32, B), blk, 0, stream>>>(
        Vb, Vt, N, D, (long long)N * D, (long long)N * D);

    // out_pre[b,i,d] = sum_j P[b,i,j] * Vt[b,d,j]   (bf16 out, overwrites Kb)
    const dim3 gop(D / 256, N / 128, B);
    gemm_nt_bf16<1><<<gop, blk, 0, stream>>>(Pb, Vt, Ob, N, D, N, N, N, D,
                                             (long long)N * N, (long long)N * D,
                                             (long long)N * D);

    // out = out_pre @ Wo^T  -> d_out head (fp32)
    gemm_nt_bf16<0><<<gproj, blk, 0, stream>>>(Ob, Wob, out_f, B * N, D, D, D, D, D, 0, 0, 0);
}
